// PointConv_42099269435604
// MI455X (gfx1250) — matrix operand order repaired and checked by reading, not compiled, not run on hardware
//
#include <hip/hip_runtime.h>

typedef __attribute__((ext_vector_type(16))) _Float16 v16h;
typedef __attribute__((ext_vector_type(8)))  float    v8f;

#define BB   4
#define NNP  2048
#define KKN  32
#define CINC 16
#define COUTC 32
#define HHC  32
#define EPSF 1e-5f
#define ROWS (BB*NNP*KKN)   /* 262144 */
#define BNC  (BB*NNP)       /* 8192   */

// ---------------------------------------------------------------- KNN --------
__global__ __launch_bounds__(128) void kknn(const float* __restrict__ xyz,
                                            int* __restrict__ idxOut) {
  __shared__ float xs[NNP * 3];        // 24 KB: whole batch xyz
  __shared__ float ds[4][NNP];         // 32 KB: one dist row per wave
  const int t = threadIdx.x, lane = t & 31, w = t >> 5;
  const int b = blockIdx.y;
  for (int i = t; i < NNP * 3; i += 128) xs[i] = xyz[b * NNP * 3 + i];
  __syncthreads();

  const int m = blockIdx.x * 4 + w;
  const float qx = xs[m * 3 + 0], qy = xs[m * 3 + 1], qz = xs[m * 3 + 2];
  for (int j = 0; j < NNP / 32; ++j) {
    int n = j * 32 + lane;
    float dx = xs[n * 3 + 0] - qx;
    float dy = xs[n * 3 + 1] - qy;
    float dz = xs[n * 3 + 2] - qz;
    ds[w][n] = dx * dx + dy * dy + dz * dz;
  }
  __syncthreads();

  const int outBase = (b * NNP + m) * KKN;
  for (int r = 0; r < KKN; ++r) {
    float best = 3.4e38f; int bi = 0x7fffffff;
    for (int j = 0; j < NNP / 32; ++j) {
      int n = j * 32 + lane;
      float v = ds[w][n];
      if (v < best) { best = v; bi = n; }
    }
    // wave-wide argmin (deterministic tie-break on smaller index)
    for (int off = 16; off; off >>= 1) {
      float ov = __shfl_xor(best, off, 32);
      int   oi = __shfl_xor(bi,   off, 32);
      if (ov < best || (ov == best && oi < bi)) { best = ov; bi = oi; }
    }
    if (lane == 0) idxOut[outBase + r] = bi;
    if (lane == (bi & 31)) ds[w][bi] = 3.4e38f;   // remove winner
    __syncthreads();
  }
}

// ------------------------------------------------------------- layer 1 ------
__global__ __launch_bounds__(256) void klayer1(const float* __restrict__ xyz,
    const int* __restrict__ idxb, const float* __restrict__ W1,
    const float* __restrict__ b1, _Float16* __restrict__ h1,
    float* __restrict__ part) {
  __shared__ float shs[8][HHC];
  __shared__ float shq[8][HHC];
  const int t = threadIdx.x, lane = t & 31, w = t >> 5;
  const int r = blockIdx.x * 256 + t;
  const int b = r / (NNP * KKN);
  const int rem = r - b * NNP * KKN;
  const int n = rem / KKN;
  const int nb = idxb[r];
  const float* xc = xyz + (size_t)(b * NNP + n)  * 3;
  const float* xn = xyz + (size_t)(b * NNP + nb) * 3;
  const float rx = xn[0] - xc[0], ry = xn[1] - xc[1], rz = xn[2] - xc[2];

  for (int o = 0; o < HHC; ++o) {
    float v = b1[o] + rx * W1[o] + ry * W1[HHC + o] + rz * W1[2 * HHC + o];
    v = v > 0.f ? v : 0.f;
    h1[(size_t)r * HHC + o] = (_Float16)v;
    float s = v, q = v * v;
    for (int off = 16; off; off >>= 1) {
      s += __shfl_xor(s, off, 32);
      q += __shfl_xor(q, off, 32);
    }
    if (lane == 0) { shs[w][o] = s; shq[w][o] = q; }
  }
  __syncthreads();
  if (t < 64) {                       // fixed-order combine across 8 waves
    int o = t & 31; bool sq = t >= 32;
    float acc = 0.f;
    for (int ww = 0; ww < 8; ++ww) acc += sq ? shq[ww][o] : shs[ww][o];
    part[blockIdx.x * 64 + t] = acc;
  }
}

// ----------------------------------------------------- deterministic reduce -
__global__ __launch_bounds__(256) void kreduce(const float* __restrict__ part,
                                               int nwg, float* __restrict__ out) {
  __shared__ float sh[4][64];
  const int t = threadIdx.x, col = t & 63, chunk = t >> 6;
  float s = 0.f;
  for (int i = chunk; i < nwg; i += 4) s += part[i * 64 + col];
  sh[chunk][col] = s;
  __syncthreads();
  if (t < 64) out[t] = sh[0][t] + sh[1][t] + sh[2][t] + sh[3][t];
}

// ------------------------------------------------------------- layer 2 ------
__global__ __launch_bounds__(256) void klayer2(const _Float16* __restrict__ h1,
    const float* __restrict__ stats1, const float* __restrict__ g1,
    const float* __restrict__ be1, const float* __restrict__ W2,
    const float* __restrict__ b2, _Float16* __restrict__ h2,
    float* __restrict__ part) {
  __shared__ _Float16 w2s[HHC * HHC];
  __shared__ _Float16 h1s[128 * HHC];
  __shared__ float a1[HHC], d1[HHC], b2s[HHC];
  __shared__ float shs[8][HHC], shq[8][HHC];
  const int t = threadIdx.x, lane = t & 31, w = t >> 5;
  const int lanelo = lane & 15, hi = lane >> 4;

  if (t < HHC) {     // fold BN1 into scale/shift
    float mean = stats1[t] * (1.f / ROWS);
    float var  = stats1[32 + t] * (1.f / ROWS) - mean * mean;
    float rstd = rsqrtf(var + EPSF);
    float a = g1[t] * rstd;
    a1[t] = a; d1[t] = be1[t] - a * mean; b2s[t] = b2[t];
  }
  for (int i = t; i < HHC * HHC; i += 256) w2s[i] = (_Float16)W2[i];
  __syncthreads();

  const int rowbase = blockIdx.x * 128;
  for (int e = t * 16; e < t * 16 + 16; ++e) {       // stage normalized h1 (f16)
    int rl = e >> 5, c = e & 31;
    float v = a1[c] * (float)h1[(size_t)(rowbase + rl) * HHC + c] + d1[c];
    h1s[e] = (_Float16)v;
  }
  __syncthreads();

  v16h A;
#pragma unroll
  for (int j = 0; j < 16; ++j) {
    int k = j + 8 * (j >> 3) + 8 * hi;               // ISA 16-bit A 16x32 layout
    A[j] = h1s[(w * 16 + lanelo) * HHC + k];
  }
#pragma unroll
  for (int nt = 0; nt < 2; ++nt) {
    v16h Bv;
#pragma unroll
    for (int j = 0; j < 16; ++j) {                   // ISA 16-bit B 32x16 layout
      int k = j + 8 * (j >> 3) + 8 * hi;
      Bv[j] = w2s[k * HHC + nt * 16 + lanelo];
    }
    v8f C;
    float bias = b2s[nt * 16 + lanelo];
#pragma unroll
    for (int g = 0; g < 8; ++g) C[g] = bias;
    v8f D = __builtin_amdgcn_wmma_f32_16x16x32_f16(false, A, false, Bv,
                                                   (short)0, C, false, false);
    float s = 0.f, q = 0.f;
#pragma unroll
    for (int g = 0; g < 8; ++g) {
      float v = D[g]; v = v > 0.f ? v : 0.f;
      int row = rowbase + w * 16 + g + 8 * hi;
      h2[(size_t)row * HHC + nt * 16 + lanelo] = (_Float16)v;
      s += v; q += v * v;
    }
    s += __shfl_xor(s, 16, 32);
    q += __shfl_xor(q, 16, 32);
    if (hi == 0) { shs[w][nt * 16 + lanelo] = s; shq[w][nt * 16 + lanelo] = q; }
  }
  __syncthreads();
  if (t < 64) {
    int o = t & 31; bool sq = t >= 32;
    float acc = 0.f;
    for (int ww = 0; ww < 8; ++ww) acc += sq ? shq[ww][o] : shs[ww][o];
    part[blockIdx.x * 64 + t] = acc;
  }
}

// --------------------- layer 3 fused with points-contraction + max-over-K ---
__global__ __launch_bounds__(128) void klayer3(const _Float16* __restrict__ h2,
    const float* __restrict__ stats2, const float* __restrict__ g2,
    const float* __restrict__ be2, const float* __restrict__ W3,
    const float* __restrict__ b3, const float* __restrict__ points,
    const int* __restrict__ idxb, float* __restrict__ outraw,
    float* __restrict__ part) {
  __shared__ _Float16 w3s[HHC * 512];     // 32 KB
  __shared__ _Float16 h2s[KKN * HHC];
  __shared__ float ptsS[KKN * CINC];
  __shared__ float b3s[512];
  __shared__ float a2[HHC], d2[HHC];
  __shared__ float red[2 * COUTC];
  const int t = threadIdx.x, lane = t & 31, w = t >> 5;
  const int lanelo = lane & 15, hi = lane >> 4;
  const int n = blockIdx.x, b = blockIdx.y;

  if (t < HHC) {
    float mean = stats2[t] * (1.f / ROWS);
    float var  = stats2[32 + t] * (1.f / ROWS) - mean * mean;
    float rstd = rsqrtf(var + EPSF);
    float a = g2[t] * rstd;
    a2[t] = a; d2[t] = be2[t] - a * mean;
  }
  for (int i = t; i < HHC * 512; i += 128) w3s[i] = (_Float16)W3[i];
  for (int i = t; i < 512; i += 128) b3s[i] = b3[i];
  __syncthreads();

  const int r0 = (b * NNP + n) * KKN;
  for (int e = t; e < KKN * HHC; e += 128) {          // normalized h2 (f16)
    int c = e & 31;
    h2s[e] = (_Float16)(a2[c] * (float)h2[(size_t)r0 * HHC + e] + d2[c]);
  }
  for (int e = t; e < KKN * CINC; e += 128) {          // gather neighbor points
    int k = e >> 4, c = e & 15;
    int nb = idxb[r0 + k];
    ptsS[e] = points[(size_t)(b * NNP + nb) * CINC + c];
  }
  __syncthreads();

  const int mtile = w >> 1, oh = w & 1;    // wave owns (k-half, out-half) tile
  v16h A;
#pragma unroll
  for (int j = 0; j < 16; ++j) {
    int k = j + 8 * (j >> 3) + 8 * hi;
    A[j] = h2s[(mtile * 16 + lanelo) * HHC + k];
  }
  float acc[8];
#pragma unroll
  for (int g = 0; g < 8; ++g) acc[g] = 0.f;

#pragma unroll
  for (int c = 0; c < CINC; ++c) {         // 16 WMMAs per wave, fused with pts
    int nt = c * 2 + oh;                   // global col = c*32 + oh*16 + lanelo
    v16h Bv;
#pragma unroll
    for (int j = 0; j < 16; ++j)
      Bv[j] = w3s[(j + 8 * (j >> 3) + 8 * hi) * 512 + nt * 16 + lanelo];
    v8f C = {};
    v8f D = __builtin_amdgcn_wmma_f32_16x16x32_f16(false, A, false, Bv,
                                                   (short)0, C, false, false);
    float bc = b3s[nt * 16 + lanelo];
#pragma unroll
    for (int g = 0; g < 8; ++g) {
      int k = mtile * 16 + g + 8 * hi;
      acc[g] += ptsS[k * CINC + c] * (D[g] + bc);
    }
  }
  // max over this wave's 16 k values
  float mx = acc[0];
#pragma unroll
  for (int g = 1; g < 8; ++g) mx = acc[g] > mx ? acc[g] : mx;
  float om = __shfl_xor(mx, 16, 32);
  mx = om > mx ? om : mx;
  if (hi == 0) red[mtile * COUTC + oh * 16 + lanelo] = mx;
  __syncthreads();

  if (t < COUTC) {
    float v = red[t] > red[COUTC + t] ? red[t] : red[COUTC + t];
    int wg = b * NNP + n;
    outraw[(size_t)wg * COUTC + t] = v;
    part[wg * 64 + t]      = v;
    part[wg * 64 + 32 + t] = v * v;
  }
}

// ------------------------------------------------------------ final BN ------
__global__ __launch_bounds__(256) void kfinal(const float* __restrict__ outraw,
    const float* __restrict__ statsF, const float* __restrict__ gbn,
    const float* __restrict__ bbn, float* __restrict__ out) {
  const int i = blockIdx.x * 256 + threadIdx.x;      // < BN*COUT
  const int o = i & 31;
  float mean = statsF[o] * (1.f / BNC);
  float var  = statsF[32 + o] * (1.f / BNC) - mean * mean;
  float rstd = rsqrtf(var + EPSF);
  out[i] = gbn[o] * (outraw[i] - mean) * rstd + bbn[o];
}

// ---------------------------------------------------------------- driver ----
extern "C" void kernel_launch(void* const* d_in, const int* in_sizes, int n_in,
                              void* d_out, int out_size, void* d_ws, size_t ws_size,
                              hipStream_t stream) {
  const float* xyz    = (const float*)d_in[0];
  const float* points = (const float*)d_in[1];
  const float* W1  = (const float*)d_in[2];
  const float* b1  = (const float*)d_in[3];
  const float* g1  = (const float*)d_in[4];
  const float* be1 = (const float*)d_in[5];
  const float* W2  = (const float*)d_in[6];
  const float* b2  = (const float*)d_in[7];
  const float* g2  = (const float*)d_in[8];
  const float* be2 = (const float*)d_in[9];
  const float* W3  = (const float*)d_in[10];
  const float* b3  = (const float*)d_in[11];
  const float* gbn = (const float*)d_in[12];
  const float* bbn = (const float*)d_in[13];

  char* ws = (char*)d_ws;
  float*    stats1 = (float*)(ws + 0);
  float*    stats2 = (float*)(ws + 256);
  float*    statsF = (float*)(ws + 512);
  int*      idxb   = (int*)(ws + 4096);                 // 1 MB
  _Float16* h1     = (_Float16*)(ws + (size_t)( 2u << 20));  // 16 MB
  _Float16* h2     = (_Float16*)(ws + (size_t)(20u << 20));  // 16 MB
  float*    outraw = (float*)(ws + (size_t)(37u << 20));     // 1 MB
  float*    part1  = (float*)(ws + (size_t)(39u << 20));     // 256 KB
  float*    part2  = (float*)(ws + (size_t)(40u << 20));     // 512 KB
  float*    partF  = (float*)(ws + (size_t)(41u << 20));     // 2 MB
  float*    out    = (float*)d_out;

  kknn   <<<dim3(NNP / 4, BB), 128, 0, stream>>>(xyz, idxb);
  klayer1<<<ROWS / 256, 256, 0, stream>>>(xyz, idxb, W1, b1, h1, part1);
  kreduce<<<1, 256, 0, stream>>>(part1, ROWS / 256, stats1);
  klayer2<<<ROWS / 128, 256, 0, stream>>>(h1, stats1, g1, be1, W2, b2, h2, part2);
  kreduce<<<1, 256, 0, stream>>>(part2, ROWS / 128, stats2);
  klayer3<<<dim3(NNP, BB), 128, 0, stream>>>(h2, stats2, g2, be2, W3, b3,
                                             points, idxb, outraw, partF);
  kreduce<<<1, 256, 0, stream>>>(partF, BNC, statsF);
  kfinal <<<BNC * COUTC / 256, 256, 0, stream>>>(outraw, statsF, gbn, bbn, out);
}
